// TransformerBlock_1881195675713
// MI455X (gfx1250) — matrix operand order repaired and checked by reading, not compiled, not run on hardware
//
#include <hip/hip_runtime.h>
#include <hip/hip_bf16.h>
#include <math.h>

typedef __attribute__((ext_vector_type(16))) _Float16 v16h;
typedef __attribute__((ext_vector_type(8)))  _Float16 v8h;
typedef __attribute__((ext_vector_type(4)))  _Float16 v4h;
typedef __attribute__((ext_vector_type(8)))  float    v8f;
typedef __attribute__((ext_vector_type(4)))  unsigned int v4u;
typedef __attribute__((ext_vector_type(8)))  int      v8i;
typedef __attribute__((ext_vector_type(4)))  int      v4i;

#define B_   4
#define S_   2048
#define D_   1024
#define H_   16
#define AD_  64
#define EPSLN 1e-5f

// ---------------------------------------------------------------- helpers
static __device__ inline v16h pack16(v8h lo, v8h hi) {
  v16h r;
#pragma unroll
  for (int i = 0; i < 8; ++i) { r[i] = lo[i]; r[i + 8] = hi[i]; }
  return r;
}
static __device__ inline v8f zero8() {
  v8f r;
#pragma unroll
  for (int i = 0; i < 8; ++i) r[i] = 0.0f;
  return r;
}
static __device__ inline v8f wmma16(v16h a, v16h b, v8f c) {
  // D = A(16x32 f16) * B(32x16 f16) + C(16x16 f32)
  return __builtin_amdgcn_wmma_f32_16x16x32_f16(false, a, false, b, (short)0, c,
                                                false, false);
}

// -------------------------------------------------- TDM 2D tile -> LDS DMA
// Builds a Tensor DMA Descriptor (D#) per CDNA5 ISA ch.8 and issues
// TENSOR_LOAD_TO_LDS.  2D tile: tile_d0 elements per line (f16), tile_d1
// lines, line stride = stride_elems.  pad_interval/pad_amount (encoded per
// ISA: interval code c -> pad after 2^(c+1) dwords; amount code a -> a+1
// dwords) let the TDM produce a padded LDS pitch directly.
static __device__ inline void tdm_load_2d_f16(unsigned int lds_addr,
                                              const void* gptr,
                                              unsigned int tile_d0,
                                              unsigned int tile_d1,
                                              unsigned int tensor_d0,
                                              unsigned int tensor_d1,
                                              unsigned long long stride_elems,
                                              unsigned int pad_interval,
                                              unsigned int pad_amount) {
  const unsigned long long ga = (unsigned long long)(uintptr_t)gptr;
  v4u g0;
  g0[0] = 1u;                                   // count=1, user mode
  g0[1] = lds_addr;                             // LDS byte address
  g0[2] = (unsigned int)ga;                     // global_addr[31:0]
  g0[3] = (unsigned int)((ga >> 32) & 0x01FFFFFFull) | 0x80000000u; // type=2
  v8i g1;
  // data_size=1 (2 bytes), pad_enable, pad codes
  g1[0] = (int)((1u << 16) | (1u << 20) | (pad_interval << 22) | (pad_amount << 25));
  g1[1] = (int)((tensor_d0 & 0xffffu) << 16);                       // dim0 lo
  g1[2] = (int)(((tensor_d0 >> 16) & 0xffffu) | ((tensor_d1 & 0xffffu) << 16));
  g1[3] = (int)(((tensor_d1 >> 16) & 0xffffu) | (tile_d0 << 16));
  g1[4] = (int)(tile_d1 & 0xffffu);                                 // tile_d1
  g1[5] = (int)(stride_elems & 0xffffffffull);                      // stride lo
  g1[6] = (int)((stride_elems >> 32) & 0xffffull);                  // stride hi
  g1[7] = 0;
  v4i z4; z4[0] = 0; z4[1] = 0; z4[2] = 0; z4[3] = 0;  // groups 2/3 unused (2D)
  v8i z8;
#pragma unroll
  for (int i = 0; i < 8; ++i) z8[i] = 0;
  // 6-arg toolchain variant (clang-23 / therock-10.0 headers)
  __builtin_amdgcn_tensor_load_to_lds(g0, g1, z4, z4, z8, 0);
}

// ---------------------------------------------------------------- f32->f16
__global__ __launch_bounds__(256) void cvt_f16(const float* __restrict__ in,
                                               _Float16* __restrict__ out, int n) {
  int i = blockIdx.x * blockDim.x + threadIdx.x;
  int stride = gridDim.x * blockDim.x;
  for (; i < n; i += stride) out[i] = (_Float16)in[i];
}

// ---------------------------------------------------------------- layernorm
__global__ __launch_bounds__(256) void ln_rows(const float* __restrict__ x,
                                               const float* __restrict__ g,
                                               const float* __restrict__ bv,
                                               _Float16* __restrict__ out) {
  const int row  = blockIdx.x;
  const int t    = threadIdx.x;
  const float4 v = ((const float4*)(x + (size_t)row * D_))[t];
  float s  = v.x + v.y + v.z + v.w;
  float ss = v.x * v.x + v.y * v.y + v.z * v.z + v.w * v.w;
#pragma unroll
  for (int m = 1; m < 32; m <<= 1) {
    s  += __shfl_xor(s, m, 32);
    ss += __shfl_xor(ss, m, 32);
  }
  __shared__ float ps[8], pss[8];
  const int wave = t >> 5, lane = t & 31;
  if (lane == 0) { ps[wave] = s; pss[wave] = ss; }
  __syncthreads();
  if (t == 0) {
    float a = 0.f, c = 0.f;
#pragma unroll
    for (int i = 0; i < 8; ++i) { a += ps[i]; c += pss[i]; }
    ps[0] = a; pss[0] = c;
  }
  __syncthreads();
  const float mu   = ps[0] * (1.0f / D_);
  const float var  = pss[0] * (1.0f / D_) - mu * mu;
  const float rstd = rsqrtf(var + EPSLN);
  const int base = t * 4;
  const float vv[4] = {v.x, v.y, v.z, v.w};
  v4h o;
#pragma unroll
  for (int i = 0; i < 4; ++i)
    o[i] = (_Float16)((vv[i] - mu) * rstd * g[base + i] + bv[base + i]);
  *(v4h*)(out + (size_t)row * D_ + base) = o;
}

// ---------------------------------------------------------------- WMMA GEMM
// C[M,N] = A[M,K](f16) * B[K,N](f16) + bias, optional +resid, optional GELU.
// Tiles staged by the Tensor Data Mover, double-buffered, overlapped with
// WMMA compute; completion tracked with TENSORcnt.
__global__ __launch_bounds__(256) void gemm_wmma(
    const _Float16* __restrict__ A, const _Float16* __restrict__ Bm,
    const float* __restrict__ bias, const float* __restrict__ resid,
    float* __restrict__ out32, _Float16* __restrict__ out16,
    int M, int N, int K, int do_gelu) {
  // TDM pads A rows (64B) with 4 dwords -> pitch 40 halves; B rows (128B)
  // with 4 dwords -> pitch 72 halves.
  __shared__ __align__(16) _Float16 As[2][128][40];  // [m][k]
  __shared__ __align__(16) _Float16 Bs[2][32][72];   // [k][n] row-major

  const int tid  = threadIdx.x;
  const int lane = tid & 31;
  const int wave = tid >> 5;
  const int hi   = lane >> 4;   // 0/1
  const int l15  = lane & 15;
  const int wm   = wave >> 1;   // 0..3 -> 32-row strip
  const int wn   = wave & 1;    // 0..1 -> 32-col strip
  const int m0   = blockIdx.y * 128;
  const int n0   = blockIdx.x * 64;

  const unsigned int ldsA[2] = {(unsigned int)(uintptr_t)&As[0][0][0],
                                (unsigned int)(uintptr_t)&As[1][0][0]};
  const unsigned int ldsB[2] = {(unsigned int)(uintptr_t)&Bs[0][0][0],
                                (unsigned int)(uintptr_t)&Bs[1][0][0]};

  v8f acc[2][2];
#pragma unroll
  for (int i = 0; i < 2; ++i)
#pragma unroll
    for (int j = 0; j < 2; ++j) acc[i][j] = zero8();

  const int NKB = K / 32;
  // prologue: DMA first tiles into buffer 0
  if (wave == 0) {
    tdm_load_2d_f16(ldsA[0], A + (size_t)m0 * K, 32, 128,
                    (unsigned)K, (unsigned)M, (unsigned long long)K, 3, 3);
    tdm_load_2d_f16(ldsB[0], Bm + n0, 64, 32,
                    (unsigned)N, (unsigned)K, (unsigned long long)N, 4, 3);
  }

  for (int t = 0; t < NKB; ++t) {
    const int buf = t & 1;
    if (wave == 0) __builtin_amdgcn_s_wait_tensorcnt((short)0);
    __syncthreads();  // current tiles resident in LDS for all waves

    // kick DMA for next tile while we compute this one
    if (wave == 0 && t + 1 < NKB) {
      const int kk = (t + 1) * 32;
      tdm_load_2d_f16(ldsA[buf ^ 1], A + (size_t)m0 * K + kk, 32, 128,
                      (unsigned)K, (unsigned)M, (unsigned long long)K, 3, 3);
      tdm_load_2d_f16(ldsB[buf ^ 1], Bm + (size_t)kk * N + n0, 64, 32,
                      (unsigned)N, (unsigned)K, (unsigned long long)N, 4, 3);
    }

    v16h af[2], bf[2];
#pragma unroll
    for (int i = 0; i < 2; ++i) {
      const int m  = wm * 32 + i * 16 + l15;
      const int kb = hi * 16;  // A layout: k = i + 16*hi
      af[i] = pack16(*(const v8h*)&As[buf][m][kb],
                     *(const v8h*)&As[buf][m][kb + 8]);
    }
#pragma unroll
    for (int j = 0; j < 2; ++j) {
      const int n = wn * 32 + j * 16 + l15;
      v16h b;
#pragma unroll
      for (int i = 0; i < 16; ++i)        // B layout: k = i + 16*hi, col = n
        b[i] = Bs[buf][hi * 16 + i][n];
      bf[j] = b;
    }
#pragma unroll
    for (int i = 0; i < 2; ++i)
#pragma unroll
      for (int j = 0; j < 2; ++j) acc[i][j] = wmma16(af[i], bf[j], acc[i][j]);
    __syncthreads();  // done reading buf before TDM refills it next round
  }

  // epilogue: C layout row = r + 8*hi, col = lane&15
#pragma unroll
  for (int i = 0; i < 2; ++i) {
#pragma unroll
    for (int j = 0; j < 2; ++j) {
      const int n = n0 + wn * 32 + j * 16 + l15;
      const float bb = bias ? bias[n] : 0.0f;
#pragma unroll
      for (int r = 0; r < 8; ++r) {
        const int m = m0 + wm * 32 + i * 16 + r + hi * 8;
        float v = acc[i][j][r] + bb;
        if (resid) v += resid[(size_t)m * N + n];
        if (do_gelu) v = 0.5f * v * (1.0f + erff(v * 0.70710678118654752f));
        if (out32) out32[(size_t)m * N + n] = v;
        if (out16) out16[(size_t)m * N + n] = (_Float16)v;
      }
    }
  }
}

// ---------------------------------------------------------------- flash attention
// qkv[b,s, h*192 + {0,64,128} + d] (f16). ctx[b,s, h*64+d] (f16).
__global__ __launch_bounds__(128) void attn_flash(const _Float16* __restrict__ qkv,
                                                  _Float16* __restrict__ ctx) {
  __shared__ __align__(16) _Float16 Vt[64][40];     // [d][key_local]
  __shared__ __align__(16) _Float16 P[4][16][40];   // per-wave probs [q][key_local]

  const int tid  = threadIdx.x;
  const int lane = tid & 31;
  const int wave = tid >> 5;
  const int hi   = lane >> 4;
  const int l15  = lane & 15;
  const int b    = blockIdx.z;
  const int h    = blockIdx.y;
  const int q0   = blockIdx.x * 64;
  const int q0w  = q0 + wave * 16;

  const size_t rs = 3 * D_;                 // qkv row stride (3072)
  const size_t hb = (size_t)h * 3 * AD_;    // h*192

  // Q fragments (A layout), reused across all key blocks
  v16h aQ[2];
  {
    const int qr = q0w + l15;
    const _Float16* qp = qkv + ((size_t)(b * S_) + qr) * rs + hb;
#pragma unroll
    for (int k2 = 0; k2 < 2; ++k2) {
      const int kk = k2 * 32, kb = hi * 16;
      aQ[k2] = pack16(*(const v8h*)(qp + kk + kb), *(const v8h*)(qp + kk + kb + 8));
    }
  }

  v8f O[4];
#pragma unroll
  for (int j = 0; j < 4; ++j) O[j] = zero8();
  float mrow[8], lrow[8];
#pragma unroll
  for (int r = 0; r < 8; ++r) { mrow[r] = -1e30f; lrow[r] = 0.0f; }

  const int kb_end = q0 + 64;
  for (int kb = 0; kb < kb_end; kb += 32) {
    // stage V block (32 keys x 64 dims) transposed into Vt[d][key]
    {
      const int keyl = tid >> 2;           // 0..31
      const int d0   = (tid & 3) * 16;     // 0,16,32,48
      const _Float16* vp =
          qkv + ((size_t)(b * S_) + kb + keyl) * rs + hb + 2 * AD_ + d0;
      v8h v0 = *(const v8h*)vp;
      v8h v1 = *(const v8h*)(vp + 8);
      if (kb + 32 < kb_end) __builtin_prefetch(vp + 32 * rs, 0, 1);
#pragma unroll
      for (int u = 0; u < 8; ++u) {
        Vt[d0 + u][keyl]     = v0[u];
        Vt[d0 + 8 + u][keyl] = v1[u];
      }
    }
    __syncthreads();

    if (kb <= q0w + 15) {  // wave-uniform: EXEC stays all-ones for WMMA
      float sc[2][8];
#pragma unroll
      for (int t2 = 0; t2 < 2; ++t2) {
        v8f s = zero8();
        const int key = kb + t2 * 16 + l15;
        const _Float16* kp = qkv + ((size_t)(b * S_) + key) * rs + hb + AD_;
#pragma unroll
        for (int k2 = 0; k2 < 2; ++k2) {
          const int off = k2 * 32 + hi * 16;  // B layout: k = i + 16*hi
          v16h bK = pack16(*(const v8h*)(kp + off), *(const v8h*)(kp + off + 8));
          s = wmma16(aQ[k2], bK, s);
        }
#pragma unroll
        for (int r = 0; r < 8; ++r) {
          const int qr = q0w + r + hi * 8;
          sc[t2][r] = (key <= qr) ? s[r] * 0.125f : -1e30f;  // 1/sqrt(64), causal
        }
      }
      // online softmax: row stats reduced across the 16 column lanes
      float alpha[8];
#pragma unroll
      for (int r = 0; r < 8; ++r) {
        float mx = fmaxf(sc[0][r], sc[1][r]);
#pragma unroll
        for (int msk = 1; msk < 16; msk <<= 1)
          mx = fmaxf(mx, __shfl_xor(mx, msk, 32));
        const float mnew = fmaxf(mrow[r], mx);
        alpha[r] = __expf(mrow[r] - mnew);
        mrow[r]  = mnew;
        const float p0 = __expf(sc[0][r] - mnew);
        const float p1 = __expf(sc[1][r] - mnew);
        P[wave][r + hi * 8][l15]      = (_Float16)p0;
        P[wave][r + hi * 8][16 + l15] = (_Float16)p1;
        float t = p0 + p1;
#pragma unroll
        for (int msk = 1; msk < 16; msk <<= 1) t += __shfl_xor(t, msk, 32);
        lrow[r] = lrow[r] * alpha[r] + t;
      }
#pragma unroll
      for (int j = 0; j < 4; ++j)
#pragma unroll
        for (int r = 0; r < 8; ++r) O[j][r] *= alpha[r];

      // P as A fragment (C-layout -> A-layout via per-wave LDS bounce)
      v16h aP;
      {
        const int kb2 = hi * 16;
        aP = pack16(*(const v8h*)&P[wave][l15][kb2],
                    *(const v8h*)&P[wave][l15][kb2 + 8]);
      }
#pragma unroll
      for (int j = 0; j < 4; ++j) {
        const int d  = j * 16 + l15;
        const int ko = hi * 16;
        v16h bV = pack16(*(const v8h*)&Vt[d][ko], *(const v8h*)&Vt[d][ko + 8]);
        O[j] = wmma16(aP, bV, O[j]);
      }
    }
    __syncthreads();
  }

#pragma unroll
  for (int r = 0; r < 8; ++r) {
    const float linv = 1.0f / lrow[r];
    const int qr = q0w + r + hi * 8;
#pragma unroll
    for (int j = 0; j < 4; ++j) {
      const int d = j * 16 + l15;
      ctx[((size_t)(b * S_) + qr) * D_ + h * AD_ + d] = (_Float16)(O[j][r] * linv);
    }
  }
}

// ---------------------------------------------------------------- launch
extern "C" void kernel_launch(void* const* d_in, const int* in_sizes, int n_in,
                              void* d_out, int out_size, void* d_ws, size_t ws_size,
                              hipStream_t stream) {
  (void)in_sizes; (void)n_in; (void)out_size; (void)ws_size;
  const float* x     = (const float*)d_in[0];
  const float* ln1w  = (const float*)d_in[1];
  const float* ln1b  = (const float*)d_in[2];
  const float* ln2w  = (const float*)d_in[3];
  const float* ln2b  = (const float*)d_in[4];
  const float* Wqkv  = (const float*)d_in[5];
  const float* bqkv  = (const float*)d_in[6];
  const float* Wo    = (const float*)d_in[7];
  const float* bo    = (const float*)d_in[8];
  const float* Wup   = (const float*)d_in[9];
  const float* bup   = (const float*)d_in[10];
  const float* Wdown = (const float*)d_in[11];
  const float* bdown = (const float*)d_in[12];
  float* out = (float*)d_out;

  const int M = B_ * S_;  // 8192

  size_t off = 0;
  auto carve = [&](size_t bytes) -> void* {
    void* p = (char*)d_ws + off;
    off += (bytes + 255) & ~(size_t)255;
    return p;
  };
  _Float16* Wqkv16 = (_Float16*)carve((size_t)D_ * 3 * D_ * 2);
  _Float16* Wo16   = (_Float16*)carve((size_t)D_ * D_ * 2);
  _Float16* Wup16  = (_Float16*)carve((size_t)D_ * 4 * D_ * 2);
  _Float16* Wdn16  = (_Float16*)carve((size_t)4 * D_ * D_ * 2);
  _Float16* xn16   = (_Float16*)carve((size_t)M * D_ * 2);
  _Float16* qkv16  = (_Float16*)carve((size_t)M * 3 * D_ * 2);
  _Float16* ctx16  = (_Float16*)carve((size_t)M * D_ * 2);
  float*    attn_o = (float*)   carve((size_t)M * D_ * 4);
  _Float16* hn16   = (_Float16*)carve((size_t)M * D_ * 2);
  _Float16* h16    = (_Float16*)carve((size_t)M * 4 * D_ * 2);

  // 1. weights -> f16
  cvt_f16<<<2048, 256, 0, stream>>>(Wqkv,  Wqkv16, D_ * 3 * D_);
  cvt_f16<<<2048, 256, 0, stream>>>(Wo,    Wo16,   D_ * D_);
  cvt_f16<<<2048, 256, 0, stream>>>(Wup,   Wup16,  D_ * 4 * D_);
  cvt_f16<<<2048, 256, 0, stream>>>(Wdown, Wdn16,  4 * D_ * D_);

  // 2. LN1
  ln_rows<<<M, 256, 0, stream>>>(x, ln1w, ln1b, xn16);

  // 3. QKV projection: [8192,1024] x [1024,3072]
  gemm_wmma<<<dim3(3 * D_ / 64, M / 128), 256, 0, stream>>>(
      xn16, Wqkv16, bqkv, nullptr, nullptr, qkv16, M, 3 * D_, D_, 0);

  // 4. causal flash attention
  attn_flash<<<dim3(S_ / 64, H_, B_), 128, 0, stream>>>(qkv16, ctx16);

  // 5. O projection + residual(x) -> f32
  gemm_wmma<<<dim3(D_ / 64, M / 128), 256, 0, stream>>>(
      ctx16, Wo16, bo, x, attn_o, nullptr, M, D_, D_, 0);

  // 6. LN2
  ln_rows<<<M, 256, 0, stream>>>(attn_o, ln2w, ln2b, hn16);

  // 7. MLP up + exact GELU: [8192,1024] x [1024,4096]
  gemm_wmma<<<dim3(4 * D_ / 64, M / 128), 256, 0, stream>>>(
      hn16, Wup16, bup, nullptr, nullptr, h16, M, 4 * D_, D_, 1);

  // 8. MLP down + residual(attn_out) -> d_out (f32)
  gemm_wmma<<<dim3(D_ / 64, M / 128), 256, 0, stream>>>(
      h16, Wdn16, bdown, attn_o, out, nullptr, M, D_, 4 * D_, 0);
}
